// MultiHeadedAttention_75969381532035
// MI455X (gfx1250) — matrix operand order repaired and checked by reading, not compiled, not run on hardware
//
#include <hip/hip_runtime.h>
#include <hip/hip_bf16.h>

// ---------------------------------------------------------------------------
// MHA, fused bf16-WMMA pipeline for gfx1250 (MI455X).
// B=2, S=2048, D=1024, H=16, dk=64.
// Compute-bound (~69 GFLOP vs ~6us of HBM traffic at 23.3 TB/s) -> all GEMM
// work on v_wmma_f32_16x16x32_bf16 with fp32 accumulation; bf16 operands are
// staged into LDS with GLOBAL_LOAD_ASYNC_TO_LDS_B128 (ASYNCcnt) and K/V tiles
// are double-buffered so async copies overlap the WMMA pipe.
// ---------------------------------------------------------------------------

typedef __attribute__((ext_vector_type(16))) __bf16 v16bf;
typedef __attribute__((ext_vector_type(8)))  float  v8f;

union BF16Frag { v16bf v; unsigned int u[8]; };

__device__ __forceinline__ unsigned short f2bf(float x) {
  unsigned int u = __float_as_uint(x);
  u += 0x7FFFu + ((u >> 16) & 1u);           // round-to-nearest-even
  return (unsigned short)(u >> 16);
}

// Async copy 16 bytes global -> LDS (gfx1250 GLOBAL_LOAD_ASYNC_TO_LDS_B128).
// Generic LDS pointer low 32 bits == LDS offset (ISA 10.2 aperture mapping).
__device__ __forceinline__ void async_copy_b128(void* lds_dst, const void* gsrc) {
  unsigned lo = (unsigned)(unsigned long long)lds_dst;
  unsigned long long ga = (unsigned long long)gsrc;
  asm volatile("global_load_async_to_lds_b128 %0, %1, off"
               :: "v"(lo), "v"(ga) : "memory");
}
__device__ __forceinline__ void async_wait0() {
  asm volatile("s_wait_asynccnt 0x0" ::: "memory");
}

// A-matrix fragment (16x32 bf16), ISA layout:
// lane L: row = L%16, half = L/16; VGPR v holds K pair k = 2v + (v>=4?8:0) + 8*half.
__device__ __forceinline__ v16bf load_a_frag(const unsigned short* lds, int row_base,
                                             int lane, int stride) {
  BF16Frag f;
  const int row  = row_base + (lane & 15);
  const int half = lane >> 4;
#pragma unroll
  for (int v = 0; v < 8; ++v) {
    const int k = 2 * v + ((v & 4) << 1) + 8 * half;
    f.u[v] = *reinterpret_cast<const unsigned int*>(&lds[row * stride + k]);
  }
  return f.v;
}

// B-matrix fragment (32x16 bf16) from LDS stored [col][k] (K contiguous):
// lane L: col = L%16, half = L/16; VGPR v holds K pair k = 2v + (v>=4?8:0) + 8*half.
__device__ __forceinline__ v16bf load_b_frag(const unsigned short* lds, int col_base,
                                             int lane, int stride) {
  BF16Frag f;
  const int col  = col_base + (lane & 15);
  const int half = lane >> 4;
#pragma unroll
  for (int v = 0; v < 8; ++v) {
    const int k = 2 * v + ((v & 4) << 1) + 8 * half;
    f.u[v] = *reinterpret_cast<const unsigned int*>(&lds[col * stride + k]);
  }
  return f.v;
}

__device__ __forceinline__ v8f wmma_bf16(v16bf a, v16bf b, v8f c) {
  return __builtin_amdgcn_wmma_f32_16x16x32_bf16(false, a, false, b, (short)0, c,
                                                 false, false);
}

// ---------------------------------------------------------------------------
// Kernel 1: projection GEMM.  out = bf16(X[4096x1024] @ W[1024x1024] + bias)
// mode 0: out layout [B,H,S,dk]  (Q and K)
// mode 1: out layout [B,H,dk,S]  (V transposed -> PV B-fragments contiguous)
// 256 threads = 8 waves; block tile 128x128; wave tile 32x64 -> 8 WMMA/chunk.
// fp32 inputs must be converted, so staging stays on the VALU f2bf path.
// ---------------------------------------------------------------------------
__global__ __launch_bounds__(256) void proj_gemm_kernel(
    const float* __restrict__ X, const float* __restrict__ W,
    const float* __restrict__ bias, unsigned short* __restrict__ out, int mode) {
  const int m0 = blockIdx.x * 128;
  const int n0 = blockIdx.y * 128;
  const int t = threadIdx.x, lane = t & 31, w = t >> 5;
  const int wm = (w & 3) * 32;       // wave rows: wm .. wm+31 (2 A-frags)
  const int wn = (w >> 2) * 64;      // wave cols: wn .. wn+63 (4 B-frags)

  __shared__ unsigned short Asm[128 * 34];   // [row][k]
  __shared__ unsigned short Bsm[128 * 34];   // transposed: [col][k]

  v8f acc[2][4] = {};

  for (int k0 = 0; k0 < 1024; k0 += 32) {
    for (int e = t; e < 128 * 32; e += 256) {            // A tile fp32->bf16
      const int r = e >> 5, c = e & 31;
      Asm[r * 34 + c] = f2bf(X[(size_t)(m0 + r) * 1024 + (k0 + c)]);
    }
    for (int e = t; e < 32 * 128; e += 256) {            // B tile, transpose
      const int r = e >> 7, c = e & 127;
      Bsm[c * 34 + r] = f2bf(W[(size_t)(k0 + r) * 1024 + (n0 + c)]);
    }
    if (k0 + 32 < 1024) {
      __builtin_prefetch(&X[(size_t)(m0 + (t >> 1)) * 1024 + k0 + 32], 0, 0);
      __builtin_prefetch(&W[(size_t)(k0 + 32 + (t & 31)) * 1024 + n0], 0, 0);
    }
    __syncthreads();

    const v16bf a0 = load_a_frag(Asm, wm,      lane, 34);
    const v16bf a1 = load_a_frag(Asm, wm + 16, lane, 34);
#pragma unroll
    for (int n = 0; n < 4; ++n) {
      const v16bf b = load_b_frag(Bsm, wn + n * 16, lane, 34);
      acc[0][n] = wmma_bf16(a0, b, acc[0][n]);
      acc[1][n] = wmma_bf16(a1, b, acc[1][n]);
    }
    __syncthreads();
  }

  const int half = lane >> 4, cidx = lane & 15;
#pragma unroll
  for (int i = 0; i < 2; ++i)
#pragma unroll
    for (int n = 0; n < 4; ++n)
#pragma unroll
      for (int j = 0; j < 8; ++j) {
        const int m = m0 + wm + 16 * i + j + 8 * half;   // row in [B*S]
        const int ncol = n0 + wn + n * 16 + cidx;        // col in [D]
        const float val = acc[i][n][j] + bias[ncol];
        const int b = m >> 11, s = m & 2047;
        const int h = ncol >> 6, dk = ncol & 63;
        size_t idx;
        if (mode == 0) idx = ((size_t)(b * 16 + h) * 2048 + s) * 64 + dk;
        else           idx = ((size_t)(b * 16 + h) * 64 + dk) * 2048 + s;
        out[idx] = f2bf(val);
      }
}

// ---------------------------------------------------------------------------
// Kernel 2: flash attention, one (b,h) x 64-query tile per block.
// 4 waves x 16 query rows; 32-key chunks, fp32 online softmax.
// K/V chunks staged with async b128 copies, double-buffered.
// LDS strides 72/40 elements keep every b128 LDS write 16B-aligned.
// ---------------------------------------------------------------------------
__global__ __launch_bounds__(128) void attn_kernel(
    const unsigned short* __restrict__ Qh,   // [B,H,S,dk]
    const unsigned short* __restrict__ Kh,   // [B,H,S,dk]
    const unsigned short* __restrict__ VT,   // [B,H,dk,S]
    unsigned short* __restrict__ Ah) {       // [B,S,D] bf16
  const int qt = blockIdx.x;                 // query tile 0..31
  const int bh = blockIdx.y;                 // b*16+h
  const int b = bh >> 4, h = bh & 15;
  const int t = threadIdx.x, lane = t & 31, w = t >> 5;
  const int half = lane >> 4, cidx = lane & 15;

  __shared__ unsigned short Qs[64 * 72];        // [qrow][dk]
  __shared__ unsigned short Ks[2][32 * 72];     // [key][dk]  (B-frag: col=key)
  __shared__ unsigned short Vs[2][64 * 40];     // [dk][key]  (B-frag: col=dk)
  __shared__ unsigned short Ps[4][16 * 34];     // per-wave P tile [qrow][key]

  const unsigned short* Qbase = Qh + ((size_t)bh * 2048 + qt * 64) * 64;
  const unsigned short* Kbase = Kh + (size_t)bh * 2048 * 64;
  const unsigned short* Vbase = VT + (size_t)bh * 64 * 2048;

  auto stage_kv = [&](int kt, int buf) {
    for (int e = t; e < 256; e += 128) {        // K: 32 rows x 8 b128 chunks
      const int r = e >> 3, c8 = e & 7;
      async_copy_b128(&Ks[buf][r * 72 + c8 * 8],
                      &Kbase[(size_t)(kt * 32 + r) * 64 + c8 * 8]);
    }
    for (int e = t; e < 256; e += 128) {        // V^T: 64 rows x 4 b128 chunks
      const int r = e >> 2, c8 = e & 3;
      async_copy_b128(&Vs[buf][r * 40 + c8 * 8],
                      &Vbase[(size_t)r * 2048 + kt * 32 + c8 * 8]);
    }
  };

  // stage Q tile (64 rows x 8 b128 chunks) + first K/V chunk
  for (int e = t; e < 512; e += 128) {
    const int r = e >> 3, c8 = e & 7;
    async_copy_b128(&Qs[r * 72 + c8 * 8], &Qbase[(size_t)r * 64 + c8 * 8]);
  }
  stage_kv(0, 0);
  async_wait0();
  __syncthreads();

  v16bf aQ[2];
  aQ[0] = load_a_frag(Qs,      w * 16, lane, 72);   // dk 0..31
  aQ[1] = load_a_frag(Qs + 32, w * 16, lane, 72);   // dk 32..63

  v8f O[4] = {};
  float mrow[8], lrow[8];
#pragma unroll
  for (int j = 0; j < 8; ++j) { mrow[j] = -1e30f; lrow[j] = 0.0f; }

  for (int kt = 0; kt < 2048 / 32; ++kt) {
    const int cur = kt & 1;
    if (kt + 1 < 64) stage_kv(kt + 1, cur ^ 1);     // overlap copy with WMMA

    // scores S = Q K^T  (two 16-key subtiles, dk reduced in two 32-chunks)
    v8f sc[2] = {};
#pragma unroll
    for (int n = 0; n < 2; ++n) {
      sc[n] = wmma_bf16(aQ[0], load_b_frag(Ks[cur],      n * 16, lane, 72), sc[n]);
      sc[n] = wmma_bf16(aQ[1], load_b_frag(Ks[cur] + 32, n * 16, lane, 72), sc[n]);
    }

    // online softmax (fp32). C-layout: VGPR j / lane -> row = j+8*half, col = cidx.
    float alpha[8];
#pragma unroll
    for (int j = 0; j < 8; ++j) {
      const float s0 = sc[0][j] * 0.125f;            // 1/sqrt(64)
      const float s1 = sc[1][j] * 0.125f;
      float pm = fmaxf(s0, s1);
      pm = fmaxf(pm, __shfl_xor(pm, 1, 16));
      pm = fmaxf(pm, __shfl_xor(pm, 2, 16));
      pm = fmaxf(pm, __shfl_xor(pm, 4, 16));
      pm = fmaxf(pm, __shfl_xor(pm, 8, 16));
      const float nm = fmaxf(mrow[j], pm);
      const float p0 = __expf(s0 - nm);
      const float p1 = __expf(s1 - nm);
      sc[0][j] = p0; sc[1][j] = p1;
      float rs = p0 + p1;
      rs += __shfl_xor(rs, 1, 16);
      rs += __shfl_xor(rs, 2, 16);
      rs += __shfl_xor(rs, 4, 16);
      rs += __shfl_xor(rs, 8, 16);
      alpha[j] = __expf(mrow[j] - nm);
      lrow[j] = lrow[j] * alpha[j] + rs;
      mrow[j] = nm;
    }
#pragma unroll
    for (int d = 0; d < 4; ++d)
#pragma unroll
      for (int j = 0; j < 8; ++j) O[d][j] *= alpha[j];

    // P: C-layout fp32 -> bf16 A-layout via per-wave LDS round trip (DScnt)
    unsigned short* Pw = Ps[w];
#pragma unroll
    for (int n = 0; n < 2; ++n)
#pragma unroll
      for (int j = 0; j < 8; ++j) {
        const int row = j + 8 * half, col = n * 16 + cidx;
        Pw[row * 34 + col] = f2bf(sc[n][j]);
      }
    const v16bf aP = load_a_frag(Pw, 0, lane, 34);

    // O += P V  (four 16-dk subtiles)
#pragma unroll
    for (int d = 0; d < 4; ++d) {
      O[d] = wmma_bf16(aP, load_b_frag(Vs[cur], d * 16, lane, 40), O[d]);
    }

    async_wait0();            // retire next-chunk async copies
    __syncthreads();          // publish them; protect buffer reuse
  }

  // epilogue: normalize, store bf16 to [B,S,D] (head h occupies cols h*64..)
#pragma unroll
  for (int d = 0; d < 4; ++d) {
#pragma unroll
    for (int j = 0; j < 8; ++j) {
      const int row = qt * 64 + w * 16 + j + 8 * half;
      const int col = h * 64 + d * 16 + cidx;
      const float val = O[d][j] / lrow[j];
      Ah[((size_t)b * 2048 + row) * 1024 + col] = f2bf(val);
    }
  }
}

// ---------------------------------------------------------------------------
// Kernel 3: output projection.  out = fp32(Ah[4096x1024]bf16 @ Wo + bo)
// A tile is bf16 in global -> staged with async b128 copies (stride 40).
// ---------------------------------------------------------------------------
__global__ __launch_bounds__(256) void out_gemm_kernel(
    const unsigned short* __restrict__ A, const float* __restrict__ W,
    const float* __restrict__ bias, float* __restrict__ out) {
  const int m0 = blockIdx.x * 128;
  const int n0 = blockIdx.y * 128;
  const int t = threadIdx.x, lane = t & 31, w = t >> 5;
  const int wm = (w & 3) * 32;
  const int wn = (w >> 2) * 64;

  __shared__ unsigned short Asm[128 * 40];   // [row][k], 16B-aligned rows
  __shared__ unsigned short Bsm[128 * 34];   // transposed: [col][k]

  v8f acc[2][4] = {};

  for (int k0 = 0; k0 < 1024; k0 += 32) {
    for (int e = t; e < 512; e += 256) {                 // A: 128 rows x 4 b128
      const int r = e >> 2, c8 = e & 3;
      async_copy_b128(&Asm[r * 40 + c8 * 8],
                      &A[(size_t)(m0 + r) * 1024 + k0 + c8 * 8]);
    }
    for (int e = t; e < 32 * 128; e += 256) {            // W fp32 -> bf16
      const int r = e >> 7, c = e & 127;
      Bsm[c * 34 + r] = f2bf(W[(size_t)(k0 + r) * 1024 + (n0 + c)]);
    }
    if (k0 + 32 < 1024) {
      __builtin_prefetch(&W[(size_t)(k0 + 32 + (t & 31)) * 1024 + n0], 0, 0);
    }
    async_wait0();
    __syncthreads();

    const v16bf a0 = load_a_frag(Asm, wm,      lane, 40);
    const v16bf a1 = load_a_frag(Asm, wm + 16, lane, 40);
#pragma unroll
    for (int n = 0; n < 4; ++n) {
      const v16bf b = load_b_frag(Bsm, wn + n * 16, lane, 34);
      acc[0][n] = wmma_bf16(a0, b, acc[0][n]);
      acc[1][n] = wmma_bf16(a1, b, acc[1][n]);
    }
    __syncthreads();
  }

  const int half = lane >> 4, cidx = lane & 15;
#pragma unroll
  for (int i = 0; i < 2; ++i)
#pragma unroll
    for (int n = 0; n < 4; ++n)
#pragma unroll
      for (int j = 0; j < 8; ++j) {
        const int m = m0 + wm + 16 * i + j + 8 * half;
        const int ncol = n0 + wn + n * 16 + cidx;
        out[(size_t)m * 1024 + ncol] = acc[i][n][j] + bias[ncol];
      }
}

// ---------------------------------------------------------------------------
extern "C" void kernel_launch(void* const* d_in, const int* in_sizes, int n_in,
                              void* d_out, int out_size, void* d_ws, size_t ws_size,
                              hipStream_t stream) {
  const float* query = (const float*)d_in[0];
  const float* key   = (const float*)d_in[1];
  const float* value = (const float*)d_in[2];
  const float* Wq = (const float*)d_in[3];
  const float* bq = (const float*)d_in[4];
  const float* Wk = (const float*)d_in[5];
  const float* bk = (const float*)d_in[6];
  const float* Wv = (const float*)d_in[7];
  const float* bv = (const float*)d_in[8];
  const float* Wo = (const float*)d_in[9];
  const float* bo = (const float*)d_in[10];
  float* out = (float*)d_out;

  // workspace: 4 bf16 buffers of 4096*1024 elements (8 MB each, 32 MB total)
  const size_t N = (size_t)4096 * 1024;
  unsigned short* Qh = (unsigned short*)d_ws;
  unsigned short* Kh = Qh + N;
  unsigned short* VT = Kh + N;
  unsigned short* Ah = VT + N;

  dim3 gG(32, 8), bG(256);
  proj_gemm_kernel<<<gG, bG, 0, stream>>>(query, Wq, bq, Qh, 0);
  proj_gemm_kernel<<<gG, bG, 0, stream>>>(key,   Wk, bk, Kh, 0);
  proj_gemm_kernel<<<gG, bG, 0, stream>>>(value, Wv, bv, VT, 1);
  attn_kernel<<<dim3(32, 32), 128, 0, stream>>>(Qh, Kh, VT, Ah);
  out_gemm_kernel<<<gG, bG, 0, stream>>>(Ah, Wo, bo, out);
}
